// QuantizedKVCache_59571196395521
// MI455X (gfx1250) — matrix operand order repaired and checked by reading, not compiled, not run on hardware
//
#include <hip/hip_runtime.h>
#include <math.h>

typedef __attribute__((ext_vector_type(16))) _Float16 v16h;
typedef __attribute__((ext_vector_type(8)))  float    v8f;
typedef __attribute__((ext_vector_type(4)))  int      v4i;

#define BH    64
#define SQ    8
#define SKV   4096
#define DD    128
#define TK    32
#define NTHR  256

#define KEY_A        3.0f
#define KEY_STEP     (6.0f / 7.0f)
#define KEY_INV_STEP (7.0f / 6.0f)
#define QJL_COEF     0.0097915167f   // sqrt(pi/2)/128
#define SM_SCALE     0.0883883476f   // 1/sqrt(128)

// ---- CDNA5 async global->LDS path (guarded; sync fallback keeps compile safe) ----
#if defined(__has_builtin)
#  if __has_builtin(__builtin_amdgcn_global_load_async_to_lds_b128)
#    define HAVE_ASYNC_LDS 1
#  endif
#  if __has_builtin(__builtin_amdgcn_s_wait_asynccnt)
#    define HAVE_WAIT_ASYNC 1
#  endif
#endif

__device__ __forceinline__ void cp_tile_b128(const void* g, void* l) {
#if defined(HAVE_ASYNC_LDS)
  typedef __attribute__((address_space(1))) v4i gv4i;
  typedef __attribute__((address_space(3))) v4i lv4i;
  // Generic->AS1 is a value-preserving 64-bit cast; generic->AS3 truncates to
  // the 32-bit LDS offset (AMDGPU generic LDS pointers carry the offset in the
  // low 32 bits).
  __builtin_amdgcn_global_load_async_to_lds_b128(
      (gv4i*)(unsigned long long)(size_t)g,
      (lv4i*)(unsigned)(size_t)l, 0, 0);
#else
  *(float4*)l = *(const float4*)g;
#endif
}

template <int N>
__device__ __forceinline__ void wait_async() {
#if defined(HAVE_ASYNC_LDS)
#  if defined(HAVE_WAIT_ASYNC)
  __builtin_amdgcn_s_wait_asynccnt(N);
#  else
  asm volatile("s_wait_asynccnt %0" :: "i"(N) : "memory");
#  endif
#endif
}

struct alignas(16) SMem {
  _Float16 projT[DD][DD];       // proj^T, [mcol][d]            32 KB
  _Float16 qA[16][DD];          // Q padded to 16 rows           4 KB
  _Float16 qpA[16][DD];         // (Q@proj) padded               4 KB
  _Float16 khat[TK][DD];        // quantized keys                8 KB
  _Float16 resid[TK][DD];       // key residuals                 8 KB
  _Float16 signs[TK][DD];       // ±coef*res_norm[key]           8 KB
  _Float16 vhatT[DD][TK];       // dequantized V, transposed     8 KB
  _Float16 pT[16][TK];          // softmax probs                 1 KB
  float    sT[16][TK];          // scores                        2 KB
  float    O[16][DD];           // output accumulator            8 KB
  float    qf[SQ][DD];          // Q f32                         4 KB
  float    kraw[2][TK][DD];     // async K staging (dbl buf)    32 KB
  float    vraw[2][TK][DD];     // async V staging (dbl buf)    32 KB
  float    coefrn[TK];
  float    mstate[16];
  float    lstate[16];
  float    alpha[16];
};

__device__ __forceinline__ v8f wmma_f16(v16h a, v16h b, v8f c) {
  return __builtin_amdgcn_wmma_f32_16x16x32_f16(
      false, a, false, b, (short)0, c, false, false);
}

// A-matrix fragment, 16-bit 16x32 layout.
__device__ __forceinline__ v16h frag_A(const _Float16* base, int ld, int m0, int kc) {
  const int lane = threadIdx.x & 31;
  const _Float16* p = base + (size_t)(m0 + (lane & 15)) * ld + kc + ((lane >> 4) << 4);
  v16h a;
#pragma unroll
  for (int i = 0; i < 16; ++i) a[i] = p[i];
  return a;
}

// B-matrix fragment from transposed row-major buffer: base[n][k] == B[k][n].
__device__ __forceinline__ v16h frag_B(const _Float16* base, int ld, int n0, int kc) {
  const int lane = threadIdx.x & 31;
  const _Float16* p = base + (size_t)(n0 + (lane & 15)) * ld + kc + ((lane >> 4) << 4);
  v16h b;
#pragma unroll
  for (int i = 0; i < 16; ++i) b[i] = p[i];
  return b;
}

__global__ __launch_bounds__(NTHR)
void qkv_attn_kernel(const float* __restrict__ query,
                     const float* __restrict__ keys,
                     const float* __restrict__ values,
                     const float* __restrict__ proj,
                     float* __restrict__ out,
                     float* __restrict__ partial,
                     int ns) {
  __shared__ SMem sm;
  const int tid  = threadIdx.x;
  const int lane = tid & 31;
  const int wave = tid >> 5;
  const int bh    = blockIdx.x / ns;
  const int split = blockIdx.x % ns;
  const int chunk = SKV / ns;
  const int k0    = split * chunk;
  const int ntiles = chunk / TK;

  const float* qbase = query  + (size_t)bh * SQ  * DD;
  const float* kbase = keys   + (size_t)bh * SKV * DD;
  const float* vbase = values + (size_t)bh * SKV * DD;

  // Issue async copy of tile 0 first so HBM latency hides behind setup work.
  {
    const char* kg = (const char*)(kbase + (size_t)k0 * DD);
    const char* vg = (const char*)(vbase + (size_t)k0 * DD);
    char* kl = (char*)&sm.kraw[0][0][0];
    char* vl = (char*)&sm.vraw[0][0][0];
#pragma unroll
    for (int j = 0; j < 4; ++j) {
      cp_tile_b128(kg + tid * 16 + j * 4096, kl + tid * 16 + j * 4096);
      cp_tile_b128(vg + tid * 16 + j * 4096, vl + tid * 16 + j * 4096);
    }
  }

  // ---------------- setup ----------------
  for (int idx = tid; idx < DD * DD; idx += NTHR) {
    int d = idx >> 7, mcol = idx & 127;
    sm.projT[mcol][d] = (_Float16)proj[idx];
  }
  for (int idx = tid; idx < 16 * DD; idx += NTHR) {
    int q = idx >> 7, d = idx & 127;
    float v = (q < SQ) ? qbase[q * DD + d] : 0.0f;
    if (q < SQ) sm.qf[q][d] = v;
    sm.qA[q][d] = (_Float16)v;
    sm.O[q][d] = 0.0f;
  }
  if (tid < 8 * TK) sm.pT[8 + (tid >> 5)][tid & (TK - 1)] = (_Float16)0.0f;
  if (tid < 16) { sm.mstate[tid] = -INFINITY; sm.lstate[tid] = 0.0f; sm.alpha[tid] = 1.0f; }
  __syncthreads();

  // qp = Q @ proj (rows >= SQ zero)
  for (int o = tid; o < 16 * DD; o += NTHR) {
    int q = o >> 7, mcol = o & 127;
    float acc = 0.0f;
    if (q < SQ) {
      for (int d = 0; d < DD; ++d) acc += sm.qf[q][d] * (float)sm.projT[mcol][d];
    }
    sm.qpA[q][mcol] = (_Float16)acc;
  }
  __syncthreads();

  const int row = tid >> 3;          // key row in tile: 0..31
  const int dqs = (tid & 7) << 4;    // d chunk: 0,16,...,112

  // ---------------- KV tile loop (double-buffered async pipeline) ----------------
  for (int t = 0; t < ntiles; ++t) {
    const int cur = t & 1;
    const int ks  = k0 + t * TK;

    if (t + 1 < ntiles) {  // issue async copy of next tile into the other buffer
      const char* kg = (const char*)(kbase + (size_t)(ks + TK) * DD);
      const char* vg = (const char*)(vbase + (size_t)(ks + TK) * DD);
      char* kl = (char*)&sm.kraw[cur ^ 1][0][0];
      char* vl = (char*)&sm.vraw[cur ^ 1][0][0];
#pragma unroll
      for (int j = 0; j < 4; ++j) {
        cp_tile_b128(kg + tid * 16 + j * 4096, kl + tid * 16 + j * 4096);
        cp_tile_b128(vg + tid * 16 + j * 4096, vl + tid * 16 + j * 4096);
      }
      wait_async<8>();   // in-order: leaves next tile's 8 ops in flight
    } else {
      wait_async<0>();
    }
    __syncthreads();     // all waves' portions of tile t landed in LDS

    // --- key quantization: khat, resid, coef*res_norm ---
    {
      const float* kr = &sm.kraw[cur][row][dqs];
      float sumsq = 0.0f;
#pragma unroll
      for (int i = 0; i < 16; ++i) {
        float kv = kr[i];
        float qi = rintf((kv + KEY_A) * KEY_INV_STEP);
        qi = fminf(fmaxf(qi, 0.0f), 7.0f);
        float kh = qi * KEY_STEP - KEY_A;
        float rs = kv - kh;
        sm.khat[row][dqs + i]  = (_Float16)kh;
        sm.resid[row][dqs + i] = (_Float16)rs;
        sumsq += rs * rs;
      }
      sumsq += __shfl_xor(sumsq, 1, 32);
      sumsq += __shfl_xor(sumsq, 2, 32);
      sumsq += __shfl_xor(sumsq, 4, 32);
      if ((tid & 7) == 0) sm.coefrn[row] = sqrtf(sumsq) * QJL_COEF;
    }

    // --- value quantize/dequantize -> vhatT ---
    {
      const float* vr = &sm.vraw[cur][row][dqs];
      float vloc[16];
      float mn = INFINITY, mx = -INFINITY;
#pragma unroll
      for (int i = 0; i < 16; ++i) {
        vloc[i] = vr[i];
        mn = fminf(mn, vloc[i]);
        mx = fmaxf(mx, vloc[i]);
      }
      mn = fminf(mn, __shfl_xor(mn, 1, 32));
      mn = fminf(mn, __shfl_xor(mn, 2, 32));
      mn = fminf(mn, __shfl_xor(mn, 4, 32));
      mx = fmaxf(mx, __shfl_xor(mx, 1, 32));
      mx = fmaxf(mx, __shfl_xor(mx, 2, 32));
      mx = fmaxf(mx, __shfl_xor(mx, 4, 32));
      float rng  = fmaxf(mx - mn, 1e-8f);
      float inv  = 15.0f / rng;
      float dq15 = (mx - mn) * (1.0f / 15.0f);
#pragma unroll
      for (int i = 0; i < 16; ++i) {
        float qi = fminf(fmaxf(rintf((vloc[i] - mn) * inv), 0.0f), 15.0f);
        sm.vhatT[dqs + i][row] = (_Float16)(mn + qi * dq15);
      }
    }
    __syncthreads();

    // --- signs GEMM: S = resid(32x128) @ proj(128x128); store ±coef*rn[key] ---
#pragma unroll
    for (int mt = 0; mt < 2; ++mt) {
      int nt = wave;
      v8f acc = {};
#pragma unroll
      for (int kc = 0; kc < DD; kc += 32)
        acc = wmma_f16(frag_A(&sm.resid[0][0], DD, mt * 16, kc),
                       frag_B(&sm.projT[0][0], DD, nt * 16, kc), acc);
      int col = nt * 16 + (lane & 15);
      int mb  = mt * 16 + ((lane >> 4) << 3);
#pragma unroll
      for (int r = 0; r < 8; ++r) {
        float v = acc[r];
        float c = sm.coefrn[mb + r];
        sm.signs[mb + r][col] = (_Float16)(v > 0.0f ? c : (v < 0.0f ? -c : 0.0f));
      }
    }
    __syncthreads();

    // --- scores: Q@khat^T + qp@signs'^T, one K=256 accumulation chain ---
    if (wave < 2) {
      int nt = wave;
      v8f acc = {};
#pragma unroll
      for (int kc = 0; kc < DD; kc += 32)
        acc = wmma_f16(frag_A(&sm.qA[0][0], DD, 0, kc),
                       frag_B(&sm.khat[0][0], DD, nt * 16, kc), acc);
#pragma unroll
      for (int kc = 0; kc < DD; kc += 32)
        acc = wmma_f16(frag_A(&sm.qpA[0][0], DD, 0, kc),
                       frag_B(&sm.signs[0][0], DD, nt * 16, kc), acc);
      int col = nt * 16 + (lane & 15);
      int mb  = (lane >> 4) << 3;
#pragma unroll
      for (int r = 0; r < 8; ++r)
        sm.sT[mb + r][col] = acc[r] * SM_SCALE;
    }
    __syncthreads();

    // --- online softmax over this tile (rows 0..7) ---
    if (tid < SQ) {
      float mold = sm.mstate[tid];
      float tmax = -INFINITY;
      for (int j = 0; j < TK; ++j) tmax = fmaxf(tmax, sm.sT[tid][j]);
      float mnew = fmaxf(mold, tmax);
      float al = __expf(mold - mnew);
      float l = sm.lstate[tid] * al;
      for (int j = 0; j < TK; ++j) {
        float p = __expf(sm.sT[tid][j] - mnew);
        l += p;
        sm.pT[tid][j] = (_Float16)p;
      }
      sm.mstate[tid] = mnew;
      sm.lstate[tid] = l;
      sm.alpha[tid] = al;
    }
    __syncthreads();

    // --- rescale accumulator ---
    for (int o = tid; o < SQ * DD; o += NTHR)
      sm.O[o >> 7][o & 127] *= sm.alpha[o >> 7];
    __syncthreads();

    // --- PV: O += P(16x32) @ vhat(32x128); each wave owns one 16x16 D-tile ---
    {
      int n0  = wave * 16;
      int col = n0 + (lane & 15);
      int mb  = (lane >> 4) << 3;
      v8f c;
#pragma unroll
      for (int r = 0; r < 8; ++r) c[r] = sm.O[mb + r][col];
      c = wmma_f16(frag_A(&sm.pT[0][0], TK, 0, 0),
                   frag_B(&sm.vhatT[0][0], TK, n0, 0), c);
#pragma unroll
      for (int r = 0; r < 8; ++r) sm.O[mb + r][col] = c[r];
    }
    __syncthreads();
  }

  // ---------------- epilogue ----------------
  if (ns == 1) {
    float* ob = out + (size_t)bh * SQ * DD;
    for (int o = tid; o < SQ * DD; o += NTHR)
      ob[o] = sm.O[o >> 7][o & 127] / sm.lstate[o >> 7];
  } else {
    float* pb = partial + (size_t)blockIdx.x * (SQ * DD + 2 * SQ);
    for (int o = tid; o < SQ * DD; o += NTHR)
      pb[o] = sm.O[o >> 7][o & 127];
    if (tid < SQ) {
      pb[SQ * DD + tid]      = sm.mstate[tid];
      pb[SQ * DD + SQ + tid] = sm.lstate[tid];
    }
  }
}

// Combine per-split partials: log-sum-exp merge of (O, m, l)
__global__ __launch_bounds__(128)
void qkv_combine_kernel(const float* __restrict__ partial,
                        float* __restrict__ out, int ns) {
  const int bh  = blockIdx.x;
  const int tid = threadIdx.x;
  const int STRIDE = SQ * DD + 2 * SQ;
  const float* base = partial + (size_t)bh * ns * STRIDE;

  __shared__ float mq[SQ], lq[SQ];
  if (tid < SQ) {
    float m = -INFINITY;
    for (int s = 0; s < ns; ++s)
      m = fmaxf(m, base[s * STRIDE + SQ * DD + tid]);
    float l = 0.0f;
    for (int s = 0; s < ns; ++s)
      l += base[s * STRIDE + SQ * DD + SQ + tid] *
           __expf(base[s * STRIDE + SQ * DD + tid] - m);
    mq[tid] = m;
    lq[tid] = l;
  }
  __syncthreads();

  for (int o = tid; o < SQ * DD; o += 128) {
    int q = o >> 7;
    float acc = 0.0f;
    for (int s = 0; s < ns; ++s)
      acc += base[s * STRIDE + o] *
             __expf(base[s * STRIDE + SQ * DD + q] - mq[q]);
    out[(size_t)bh * SQ * DD + o] = acc / lq[q];
  }
}

extern "C" void kernel_launch(void* const* d_in, const int* in_sizes, int n_in,
                              void* d_out, int out_size, void* d_ws, size_t ws_size,
                              hipStream_t stream) {
  const float* query  = (const float*)d_in[0];
  const float* keys   = (const float*)d_in[1];
  const float* values = (const float*)d_in[2];
  const float* proj   = (const float*)d_in[3];
  float* out = (float*)d_out;
  float* ws  = (float*)d_ws;

  const size_t need = (size_t)BH * 4 * (SQ * DD + 2 * SQ) * sizeof(float);
  const int ns = (ws != nullptr && ws_size >= need) ? 4 : 1;

  qkv_attn_kernel<<<BH * ns, NTHR, 0, stream>>>(query, keys, values, proj, out, ws, ns);
  if (ns > 1)
    qkv_combine_kernel<<<BH, 128, 0, stream>>>(ws, out, ns);
}
